// selective_matching_crossview_20280835572215
// MI455X (gfx1250) — matrix operand order repaired and checked by reading, not compiled, not run on hardware
//
#include <hip/hip_runtime.h>
#include <hip/hip_bf16.h>
#include <stdint.h>

// ---------------------------------------------------------------------------
// Selective matching + cross-view aggregation, fused for MI455X (gfx1250).
// N=2, AN2=25, c=64, H=W=64, PS=4, pn=16 -> B=512 patches, D=1024, K=6, 9 shifts.
// GEMM stages on v_wmma_f32_16x16x32_bf16; bf16 global->LDS copies use
// global_load_async_to_lds_b128 (ASYNCcnt); operand fetch software-pipelined.
// ---------------------------------------------------------------------------

typedef __attribute__((ext_vector_type(16))) __bf16 v16bf;
typedef __attribute__((ext_vector_type(8)))  float  v8f;

static __device__ __forceinline__ unsigned short f2bf(float f) {
  unsigned int u = __float_as_uint(f);
  u += 0x7FFFu + ((u >> 16) & 1u);          // round-to-nearest-even
  return (unsigned short)(u >> 16);
}
static __device__ __forceinline__ float bf2f(unsigned int hb) {
  return __uint_as_float(hb << 16);
}
static __device__ __forceinline__ float sq8(uint4 u) {
  float s = 0.f;
  unsigned int w[4] = {u.x, u.y, u.z, u.w};
#pragma unroll
  for (int i = 0; i < 4; ++i) {
    float a = bf2f(w[i] & 0xFFFFu), b = bf2f(w[i] >> 16);
    s += a * a + b * b;
  }
  return s;
}
static __device__ __forceinline__ v16bf mk16(uint4 lo, uint4 hi) {
  union { uint4 q[2]; v16bf v; } u;
  u.q[0] = lo; u.q[1] = hi;
  return u.v;
}
static __device__ __forceinline__ float lrelu(float v) { return v > 0.f ? v : 0.1f * v; }

// Async bulk copy global(bf16)->LDS, 16B per lane-op, tracked by ASYNCcnt.
// lds_off = group-segment byte offset (low 32 bits of a generic LDS pointer).
static __device__ __forceinline__ void async_copy_b128(unsigned lds_off, const void* gsrc) {
  asm volatile("global_load_async_to_lds_b128 %0, %1, off"
               :: "v"(lds_off), "v"((unsigned long long)(uintptr_t)gsrc)
               : "memory");
}
static __device__ __forceinline__ void wait_async0() {
  asm volatile("s_wait_asynccnt 0x0" ::: "memory");
}

struct Frag { uint4 al, ah, bl, bh; };

// ---------------------------------------------------------------------------
// Kernel 0a: weights -> bf16, K-order pre-swizzled for contiguous operand loads
// ---------------------------------------------------------------------------
__global__ void k_prep_weights(const float* __restrict__ w1, const float* __restrict__ w2,
                               unsigned short* __restrict__ w1b, unsigned short* __restrict__ w2b) {
  int g = blockIdx.x * 256 + threadIdx.x;
  if (g < 64 * 384) w1b[g] = f2bf(w1[g]);
  int e = g - 64 * 384;
  if (e >= 0 && e < 64 * 1152) {
    int co = e / 1152, r = e - co * 1152;
    int tap = r >> 7, ci = r & 127;
    int ky = tap / 3, kx = tap - ky * 3;
    w2b[e] = f2bf(w2[((co * 128 + ci) * 3 + ky) * 3 + kx]);
  }
}

// ---------------------------------------------------------------------------
// Kernel 0b: lf_fea (img,ci,y,x) f32 -> lf_t (img,y,x,ci) bf16
// ---------------------------------------------------------------------------
__global__ void k_transpose_lf(const float* __restrict__ lf, unsigned short* __restrict__ lft) {
  int t = blockIdx.x * 256 + threadIdx.x;          // 50*4096 threads
  int img = t >> 12, p = t & 4095;
  const float* src = lf + (size_t)img * 64 * 4096 + p;
  unsigned short* dst = lft + (size_t)t * 64;
  for (int ci = 0; ci < 64; ++ci) dst[ci] = f2bf(src[(size_t)ci * 4096]);
}

// ---------------------------------------------------------------------------
// Kernel 1: per-patch 9x shifted Gram matrices via WMMA + top-6 selection.
// 128 threads (4 waves), one 16x16 tile each of the padded 32x32 Gram.
// Shifted views stream through a ping-pong LDS buffer (1 barrier/chunk);
// operand fragments double-buffered around the WMMA.
// ---------------------------------------------------------------------------
__global__ void k_match_topk(const float* __restrict__ lf, int* __restrict__ idxo) {
  extern __shared__ unsigned char sm[];
  unsigned short* sA   = (unsigned short*)sm;                    // 32*1024 bf16 (65536B)
  unsigned short* sB0  = (unsigned short*)(sm + 65536);          // 32*256 bf16 x2 (32768B)
  float* x2   = (float*)(sm + 65536 + 32768);                    // 32
  float* y2   = x2 + 32;                                         // 32
  float* dist = y2 + 32;                                         // 25*240

  const int b = blockIdx.x;
  const int n = b >> 8, rem = b & 255, py = rem >> 4, px = rem & 15;
  const int tid = threadIdx.x, lane = tid & 31, wv = tid >> 5;
  const int tM = wv >> 1, tN = wv & 1, kh = lane >> 4;
  const float* imgbase = lf + (size_t)n * 25 * 64 * 4096;

  // Stage patch (A) in bf16; zero-pad rows 25..31.
  for (int e = tid; e < 32 * 1024; e += 128) {
    int a = e >> 10, d = e & 1023;
    float v = 0.f;
    if (a < 25) {
      int ch = d >> 4, iy = (d >> 2) & 3, ix = d & 3;
      v = imgbase[((size_t)(a * 64 + ch)) * 4096 + (py * 4 + iy) * 64 + (px * 4 + ix)];
    }
    sA[e] = f2bf(v);
  }
  for (int e = tid; e < 2 * 32 * 256; e += 128)                  // pad both buffers once
    if (((e >> 8) & 31) >= 25) sB0[e] = 0;
  __syncthreads();

  if (tid < 25) {                                                // x2 = ||patch||^2
    const uint4* rp = (const uint4*)(sA + tid * 1024);
    float s = 0.f;
    for (int q = 0; q < 128; ++q) s += sq8(rp[q]);
    x2[tid] = s;
  }

  const unsigned short* arow = sA + (tM * 16 + (lane & 15)) * 1024;
  const int bro = (tN * 16 + (lane & 15)) * 256;

  auto stage = [&](int cc, int xs, int ys, unsigned short* buf) {
    for (int e = tid; e < 25 * 256; e += 128) {
      int m = e >> 8, dl = e & 255, d = cc * 256 + dl;
      int ch = d >> 4, iy = (d >> 2) & 3, ix = d & 3;
      int gy = ((py * 4 + iy) - xs) & 63, gx = ((px * 4 + ix) - ys) & 63;
      buf[m * 256 + dl] = f2bf(imgbase[((size_t)(m * 64 + ch)) * 4096 + gy * 64 + gx]);
    }
  };
  auto ldfrag = [&](int cc, int ks, const unsigned short* bb) {
    Frag f;
    int kg = cc * 256 + ks * 32, kl = ks * 32;
    f.al = *(const uint4*)(arow + kg + kh * 8);
    f.ah = *(const uint4*)(arow + kg + 16 + kh * 8);
    f.bl = *(const uint4*)(bb + kl + kh * 8);
    f.bh = *(const uint4*)(bb + kl + 16 + kh * 8);
    return f;
  };

  for (int sh = 0; sh < 9; ++sh) {
    int xs = (sh / 3) * 2, ys = (sh % 3) * 2;                    // roll amounts
    v8f acc = {};
    float y2r = 0.f;
    stage(0, xs, ys, sB0);                                       // prologue -> buf0
    __syncthreads();
    for (int cc = 0; cc < 4; ++cc) {
      unsigned short* cur = sB0 + (cc & 1) * 32 * 256;
      if (cc < 3) stage(cc + 1, xs, ys, sB0 + ((cc + 1) & 1) * 32 * 256);
      if (tid < 25) {                                            // y2 from staged chunk
        const uint4* rp = (const uint4*)(cur + tid * 256);
        for (int q = 0; q < 32; ++q) y2r += sq8(rp[q]);
      }
      const unsigned short* bb = cur + bro;
      Frag f = ldfrag(cc, 0, bb);
#pragma unroll
      for (int ks = 0; ks < 8; ++ks) {
        Frag fn = (ks < 7) ? ldfrag(cc, ks + 1, bb) : f;         // prefetch next k-step
        acc = __builtin_amdgcn_wmma_f32_16x16x32_bf16(false, mk16(f.al, f.ah),
                                                      false, mk16(f.bl, f.bh),
                                                      (short)0, acc, false, false);
        f = fn;
      }
      __syncthreads();                                           // cur consumed, next staged
    }
    if (tid < 25) y2[tid] = y2r;
    __syncthreads();
#pragma unroll
    for (int r = 0; r < 8; ++r) {                                // C tile -> distances
      int row = tM * 16 + r + (kh << 3);
      int col = tN * 16 + (lane & 15);
      if (row < 25 && col < 25)
        dist[row * 240 + sh * 25 + col] = x2[row] + y2[col] - 2.f * acc[r];
    }
    __syncthreads();
  }

  if (tid < 25) {                                                // top-6 (ties -> lowest idx)
    float* dr = dist + tid * 240;
    for (int k = 0; k < 6; ++k) {
      float best = 3.0e38f; int bj = 0;
      for (int j = 0; j < 225; ++j) { float v = dr[j]; if (v < best) { best = v; bj = j; } }
      idxo[(b * 25 + tid) * 6 + k] = bj;
      dr[bj] = 3.0e38f;
    }
  }
}

// ---------------------------------------------------------------------------
// Kernel 2: gather 6 winners + 1x1 conv as WMMA GEMM (M=64, K=384, N=16/patch).
// W1 panel staged with async global->LDS b128 copies; split-K over 8 waves.
// ---------------------------------------------------------------------------
__global__ void k_gather_conv1(const float* __restrict__ lf, const int* __restrict__ idxi,
                               const unsigned short* __restrict__ w1b,
                               unsigned short* __restrict__ hout) {
  extern __shared__ unsigned char sm[];
  unsigned short* sW   = (unsigned short*)sm;                    // 64*384 (49152B)
  unsigned short* sG   = (unsigned short*)(sm + 49152);          // 16*384 (12288B)
  float*          sRed = (float*)(sm + 49152 + 12288);           // 4*256  (4096B)
  int*            sIdx = (int*)(sm + 49152 + 12288 + 4096);      // 8

  const int b = blockIdx.x;
  const int n = b >> 8, rem = b & 255, py = rem >> 4, px = rem & 15;
  const int tid = threadIdx.x, lane = tid & 31, wv = tid >> 5;
  const int mt = wv & 3, kh2 = wv >> 2, kh = lane >> 4;
  const float* imgbase = lf + (size_t)n * 25 * 64 * 4096;

  { // stage W1 (bf16) via async DMA into LDS; reused for 25 views
    unsigned woff = (unsigned)(uintptr_t)sW;
    for (int e = tid; e < 3072; e += 256)
      async_copy_b128(woff + e * 16, w1b + e * 8);
    wait_async0();
  }

  const unsigned short* arow = sW + (mt * 16 + (lane & 15)) * 384;
  const unsigned short* brow = sG + (lane & 15) * 384;
  auto ldfrag = [&](int s) {
    Frag f;
    int kk = (kh2 * 6 + s) * 32;
    f.al = *(const uint4*)(arow + kk + kh * 8);
    f.ah = *(const uint4*)(arow + kk + 16 + kh * 8);
    f.bl = *(const uint4*)(brow + kk + kh * 8);
    f.bh = *(const uint4*)(brow + kk + 16 + kh * 8);
    return f;
  };

  for (int a = 0; a < 25; ++a) {
    if (tid < 6) sIdx[tid] = idxi[(b * 25 + a) * 6 + tid];
    __syncthreads();
    for (int e = tid; e < 16 * 384; e += 256) {                  // gather B panel
      int pix = e / 384, k = e - pix * 384;
      int kslot = k >> 6, ci = k & 63;
      int j = sIdx[kslot], ish = j / 25, m = j - ish * 25;
      int xs = (ish / 3) * 2, ys = (ish % 3) * 2;
      int iy = pix >> 2, ix = pix & 3;
      int gy = ((py * 4 + iy) - xs) & 63, gx = ((px * 4 + ix) - ys) & 63;
      sG[pix * 384 + k] = f2bf(imgbase[((size_t)(m * 64 + ci)) * 4096 + gy * 64 + gx]);
    }
    __syncthreads();

    v8f acc = {};
    Frag f = ldfrag(0);
#pragma unroll
    for (int s = 0; s < 6; ++s) {
      Frag fn = (s < 5) ? ldfrag(s + 1) : f;
      acc = __builtin_amdgcn_wmma_f32_16x16x32_bf16(false, mk16(f.al, f.ah),
                                                    false, mk16(f.bl, f.bh),
                                                    (short)0, acc, false, false);
      f = fn;
    }
    if (kh2 == 1) {
#pragma unroll
      for (int r = 0; r < 8; ++r) {
        int rt = r + (kh << 3);
        sRed[mt * 256 + rt * 16 + (lane & 15)] = acc[r];
      }
    }
    __syncthreads();
    if (kh2 == 0) {
#pragma unroll
      for (int r = 0; r < 8; ++r) {
        int rt = r + (kh << 3);
        float v = lrelu(acc[r] + sRed[mt * 256 + rt * 16 + (lane & 15)]);
        int co = mt * 16 + rt, pix = lane & 15;
        int y = py * 4 + (pix >> 2), x = px * 4 + (pix & 3);
        hout[((((size_t)(n * 25 + a)) * 64 + y) * 64 + x) * 64 + co] = f2bf(v);
      }
    }
    __syncthreads();
  }
}

// ---------------------------------------------------------------------------
// Kernel 3: 3x3 conv over concat(lf, h) as implicit WMMA GEMM.
// Halo (18x18x128 bf16, 81KB LDS) filled with async global->LDS b128 copies
// (zeros via ds stores on the padding border). Weight A-fragments (global,
// L2-resident) software-pipelined one k-step ahead of the WMMAs.
// ---------------------------------------------------------------------------
__global__ void k_conv3x3(const unsigned short* __restrict__ lft,
                          const unsigned short* __restrict__ hin,
                          const unsigned short* __restrict__ w2b,
                          float* __restrict__ out) {
  extern __shared__ unsigned char sm[];
  unsigned short* halo = (unsigned short*)sm;                    // 324*128 bf16 (82944B)
  const int img = blockIdx.y, tile = blockIdx.x;
  const int ty0 = (tile >> 2) * 16, tx0 = (tile & 3) * 16;
  const int tid = threadIdx.x, lane = tid & 31, wv = tid >> 5;
  const int kh = lane >> 4, xl = lane & 15;

  {
    unsigned hoff = (unsigned)(uintptr_t)halo;
    for (int e = tid; e < 5184; e += 256) {                      // fill halo
      int p = e >> 4, s = e & 15;
      int yy = p / 18, xx = p - yy * 18;
      int gy = ty0 + yy - 1, gx = tx0 + xx - 1;
      if (gy >= 0 && gy < 64 && gx >= 0 && gx < 64) {
        size_t row = (((size_t)img * 64 + gy) * 64 + gx) * 64;
        const unsigned short* src = (s < 8) ? (lft + row + s * 8)
                                            : (hin + row + (s - 8) * 8);
        async_copy_b128(hoff + p * 256 + s * 16, src);
      } else {
        uint4 z = {0u, 0u, 0u, 0u};
        ((uint4*)(halo + p * 128))[s] = z;                       // zero padding border
      }
    }
    wait_async0();
  }
  __syncthreads();

  v8f acc[8];
  { v8f z = {};
#pragma unroll
    for (int i = 0; i < 8; ++i) acc[i] = z; }

  auto ldA = [&](int kstep, v16bf* A) {
    int kk = kstep * 32;
#pragma unroll
    for (int m = 0; m < 4; ++m) {
      const unsigned short* arow = w2b + (size_t)(m * 16 + xl) * 1152;
      A[m] = mk16(*(const uint4*)(arow + kk + kh * 8),
                  *(const uint4*)(arow + kk + 16 + kh * 8));
    }
  };

  v16bf Acur[4];
  ldA(0, Acur);
  for (int kstep = 0; kstep < 36; ++kstep) {
    int tap = kstep >> 2, cib = (kstep & 3) * 32;
    int dy = tap / 3, dx = tap - dy * 3;
    if (kstep + 2 < 36)
      __builtin_prefetch(w2b + (size_t)xl * 1152 + (kstep + 2) * 32, 0, 0);
    v16bf Anx[4];
    if (kstep < 35) ldA(kstep + 1, Anx);                         // pipeline global A
#pragma unroll
    for (int nt = 0; nt < 2; ++nt) {
      int yl = wv * 2 + nt;
      const unsigned short* hp = halo + ((yl + dy) * 18 + (xl + dx)) * 128 + cib + kh * 8;
      v16bf Bop = mk16(*(const uint4*)hp, *(const uint4*)(hp + 16));
#pragma unroll
      for (int m = 0; m < 4; ++m)
        acc[nt * 4 + m] = __builtin_amdgcn_wmma_f32_16x16x32_bf16(false, Acur[m],
                                                                  false, Bop, (short)0,
                                                                  acc[nt * 4 + m],
                                                                  false, false);
    }
    if (kstep < 35) {
#pragma unroll
      for (int m = 0; m < 4; ++m) Acur[m] = Anx[m];
    }
  }

#pragma unroll
  for (int nt = 0; nt < 2; ++nt) {
    int y = ty0 + wv * 2 + nt;
#pragma unroll
    for (int m = 0; m < 4; ++m) {
#pragma unroll
      for (int r = 0; r < 8; ++r) {
        int co = m * 16 + r + (kh << 3);
        out[((size_t)img * 64 + co) * 4096 + y * 64 + tx0 + xl] =
            lrelu(acc[nt * 4 + m][r]);
      }
    }
  }
}

// ---------------------------------------------------------------------------
// Host entry. Workspace layout (needs ~52.9MB):
//   [0)            idx   512*25*6 int32        (307200 B)
//   [307200)       lf_t  50*64*64*64 bf16      (26214400 B)
//   [26521600)     h     50*64*64*64 bf16      (26214400 B)
//   [52736000)     w1b   64*384 bf16           (49152 B)
//   [52785152)     w2b   64*1152 bf16          (147456 B)
// ---------------------------------------------------------------------------
extern "C" void kernel_launch(void* const* d_in, const int* in_sizes, int n_in,
                              void* d_out, int out_size, void* d_ws, size_t ws_size,
                              hipStream_t stream) {
  (void)in_sizes; (void)n_in; (void)out_size; (void)ws_size;
  const float* lf = (const float*)d_in[0];
  const float* w1 = (const float*)d_in[1];
  const float* w2 = (const float*)d_in[2];
  float* out = (float*)d_out;
  unsigned char* ws = (unsigned char*)d_ws;

  int*            idx  = (int*)(ws + 0);
  unsigned short* lft  = (unsigned short*)(ws + 307200);
  unsigned short* h    = (unsigned short*)(ws + 26521600);
  unsigned short* w1b  = (unsigned short*)(ws + 52736000);
  unsigned short* w2b  = (unsigned short*)(ws + 52785152);

  k_prep_weights<<<384, 256, 0, stream>>>(w1, w2, w1b, w2b);
  k_transpose_lf<<<800, 256, 0, stream>>>(lf, lft);

  size_t sm1 = 65536 + 32768 + 256 + 25 * 240 * 4;   // 122560 B
  k_match_topk<<<512, 128, sm1, stream>>>(lf, idx);

  size_t sm2 = 49152 + 12288 + 4096 + 32;            // 65568 B
  k_gather_conv1<<<512, 256, sm2, stream>>>(lf, idx, w1b, h);

  size_t sm3 = 324 * 128 * 2;                        // 82944 B
  k_conv3x3<<<dim3(16, 50), 256, sm3, stream>>>(lft, h, w2b, out);
}
